// VQVAE_40707700031950
// MI455X (gfx1250) — matrix operand order repaired and checked by reading, not compiled, not run on hardware
//
#include <hip/hip_runtime.h>
#include <hip/hip_bf16.h>

#define N_ROWS 16384
#define CIN    3072
#define HID    1024
#define DLAT   32
#define KCODES 512
#define COMMITF 0.25f

typedef __attribute__((ext_vector_type(16))) __bf16  v16bf;
typedef __attribute__((ext_vector_type(8)))  float   v8f;
typedef __attribute__((ext_vector_type(4)))  unsigned int u32x4;

__device__ __forceinline__ unsigned short f32_to_bf16(float f) {
  unsigned int u = __float_as_uint(f);
  unsigned int r = u + 0x7FFFu + ((u >> 16) & 1u);   // round-to-nearest-even
  return (unsigned short)(r >> 16);
}

__global__ void cvt_f32_bf16_kernel(const float* __restrict__ in,
                                    unsigned short* __restrict__ out, int n) {
  int i = blockIdx.x * blockDim.x + threadIdx.x;
  if (i < n) out[i] = f32_to_bf16(in[i]);
}

// Wt[n*K + k] = bf16(W[k*N + n]); tiled 32x32 LDS transpose, K,N multiples of 32.
__global__ __launch_bounds__(256) void transpose_f32_bf16_kernel(
    const float* __restrict__ W, unsigned short* __restrict__ Wt, int K, int N)
{
  __shared__ unsigned short tile[32][33];
  const int tx = threadIdx.x & 31;
  const int ty = threadIdx.x >> 5;               // 0..7
  const int k0 = blockIdx.y * 32;
  const int n0 = blockIdx.x * 32;
  for (int j = 0; j < 4; ++j) {
    int k = k0 + ty + j * 8;
    tile[ty + j * 8][tx] = f32_to_bf16(W[(size_t)k * N + n0 + tx]);
  }
  __syncthreads();
  for (int j = 0; j < 4; ++j) {
    int n = n0 + ty + j * 8;
    Wt[(size_t)n * K + k0 + tx] = tile[tx][ty + j * 8];
  }
}

// C = act(A[M,K](bf16) @ W + bias) with W transposed: Wt[N,K] bf16.
// act: 0 = ReLU, 1 = sigmoid (+non-temporal f32 stores).
// Block: 256 threads (8 wave32). Block tile 256x128; wave tile 64x64 (4x4 WMMA).
// Double-buffered LDS staged with CDNA5 async global->LDS b128 copies.
__global__ __launch_bounds__(256) void gemm_bf16_wmma(
    const unsigned short* __restrict__ A,
    const unsigned short* __restrict__ Wt,
    const float* __restrict__ bias,
    unsigned short* __restrict__ outb,
    float* __restrict__ outf,
    int M, int N, int K, int act)
{
  __shared__ __align__(16) unsigned short ldsA[2][256 * 32]; // [buf][row][k]
  __shared__ __align__(16) unsigned short ldsB[2][128 * 32]; // [buf][col][k]

  const int t    = threadIdx.x;
  const int lane = t & 31;
  const int wave = t >> 5;
  const int wm   = wave & 3;   // 4 bands of 64 rows
  const int wn   = wave >> 2;  // 2 bands of 64 cols
  const int rowBase = blockIdx.y * 256;
  const int colBase = blockIdx.x * 128;

  v8f acc[4][4] = {};

  // A fragment: lanes 0-15 -> M=lane, K {0..7,16..23}; lanes 16-31 -> same M, +8
  const int am    = lane & 15;
  const int koff  = (lane >> 4) * 8;
  // B fragment: col = lane&15; lanes 0-15 -> K {0..7,16..23}; lanes 16-31 -> +8
  const int khalf = (lane >> 4) * 8;
  const int bcol  = lane & 15;

  // staging: A -> each thread owns one row (64B = 4 async b128)
  //          B -> each thread owns 32B of one col (2 async b128)
  const int bc  = t >> 1;
  const int bkb = (t & 1) * 16;

  auto stageA = [&](int kt, int buf) {
    const unsigned short* g = A + (size_t)(rowBase + t) * K + kt;
    unsigned lo = (unsigned)(size_t)&ldsA[buf][t * 32];
    unsigned long long ga = (unsigned long long)(size_t)g;
    asm volatile("global_load_async_to_lds_b128 %0, %1, off"           :: "v"(lo), "v"(ga) : "memory");
    asm volatile("global_load_async_to_lds_b128 %0, %1, off offset:16" :: "v"(lo), "v"(ga) : "memory");
    asm volatile("global_load_async_to_lds_b128 %0, %1, off offset:32" :: "v"(lo), "v"(ga) : "memory");
    asm volatile("global_load_async_to_lds_b128 %0, %1, off offset:48" :: "v"(lo), "v"(ga) : "memory");
  };
  auto stageB = [&](int kt, int buf) {
    int gcol = colBase + bc;
    unsigned lo = (unsigned)(size_t)&ldsB[buf][bc * 32 + bkb];
    if (gcol < N) {
      const unsigned short* g = Wt + (size_t)gcol * K + kt + bkb;
      unsigned long long ga = (unsigned long long)(size_t)g;
      asm volatile("global_load_async_to_lds_b128 %0, %1, off"           :: "v"(lo), "v"(ga) : "memory");
      asm volatile("global_load_async_to_lds_b128 %0, %1, off offset:16" :: "v"(lo), "v"(ga) : "memory");
    } else {
      u32x4 z = {0u, 0u, 0u, 0u};
      *(u32x4*)(&ldsB[buf][bc * 32 + bkb]) = z;
      *(u32x4*)(&ldsB[buf][bc * 32 + bkb + 8]) = z;
    }
  };

  stageA(0, 0);
  stageB(0, 0);

  int buf = 0;
  for (int kt = 0; kt < K; kt += 32, buf ^= 1) {
    asm volatile("s_wait_asynccnt 0x0" ::: "memory");
    __syncthreads();
    if (kt + 32 < K) {           // prefetch next tile into the other buffer
      stageA(kt + 32, buf ^ 1);
      stageB(kt + 32, buf ^ 1);
    }

    v16bf bfrag[4];
    for (int tj = 0; tj < 4; ++tj) {
      int col = wn * 64 + tj * 16 + bcol;
      ((u32x4*)&bfrag[tj])[0] = *(const u32x4*)(&ldsB[buf][col * 32 + khalf]);
      ((u32x4*)&bfrag[tj])[1] = *(const u32x4*)(&ldsB[buf][col * 32 + khalf + 16]);
    }
    for (int ti = 0; ti < 4; ++ti) {
      v16bf afrag;
      int row = wm * 64 + ti * 16 + am;
      ((u32x4*)&afrag)[0] = *(const u32x4*)(&ldsA[buf][row * 32 + koff]);
      ((u32x4*)&afrag)[1] = *(const u32x4*)(&ldsA[buf][row * 32 + koff + 16]);
      for (int tj = 0; tj < 4; ++tj)
        acc[ti][tj] = __builtin_amdgcn_wmma_f32_16x16x32_bf16(
            false, afrag, false, bfrag[tj], (short)0, acc[ti][tj],
            false, false);
    }
  }

  // ---- epilogue: C/D layout row = r + 8*(lane/16), col = lane&15
  const int rsub = (lane >> 4) * 8;
  const int csub = lane & 15;
  for (int ti = 0; ti < 4; ++ti) {
    for (int tj = 0; tj < 4; ++tj) {
      int gcol = colBase + wn * 64 + tj * 16 + csub;
      if (gcol >= N) continue;
      float b = bias[gcol];
      for (int r = 0; r < 8; ++r) {
        int grow = rowBase + wm * 64 + ti * 16 + rsub + r;
        float v = acc[ti][tj][r] + b;
        size_t o = (size_t)grow * N + gcol;
        if (act == 0) {
          v = fmaxf(v, 0.0f);
          if (outb) outb[o] = f32_to_bf16(v);
          if (outf) outf[o] = v;
        } else {
          v = 1.0f / (1.0f + __expf(-v));
          if (outf) __builtin_nontemporal_store(v, outf + o);  // streaming output
        }
      }
    }
  }
}

// One block per row: 512 codebook distances, argmin (first-min tie-break),
// emit q_st(f32), q(bf16) for decoder, per-row sum((q-e)^2).
__global__ __launch_bounds__(256) void vq_kernel(
    const float* __restrict__ e,       // [N,32]
    const float* __restrict__ cb,      // [512,32]
    float* __restrict__ qst,           // [N,32] -> d_out region
    unsigned short* __restrict__ qb,   // [N,32] bf16
    float* __restrict__ partial)       // [N]
{
  __shared__ float se[32];
  __shared__ float smin[256];
  __shared__ int   sidx[256];
  __shared__ int   bestIdx;
  const int t = threadIdx.x;
  const int row = blockIdx.x;

  if (t < 32) se[t] = e[(size_t)row * 32 + t];
  __syncthreads();

  float best = 3.4e38f; int bi = 0;
  for (int c = t; c < KCODES; c += 256) {
    float d2 = 0.0f;
    for (int d = 0; d < 32; ++d) {
      float diff = se[d] - cb[c * 32 + d];
      d2 += diff * diff;
    }
    if (d2 < best) { best = d2; bi = c; }
  }
  smin[t] = best; sidx[t] = bi;
  __syncthreads();
  for (int s = 128; s > 0; s >>= 1) {
    if (t < s) {
      if (smin[t + s] < smin[t] ||
          (smin[t + s] == smin[t] && sidx[t + s] < sidx[t])) {
        smin[t] = smin[t + s]; sidx[t] = sidx[t + s];
      }
    }
    __syncthreads();
  }
  if (t == 0) bestIdx = sidx[0];
  __syncthreads();

  const int idx = bestIdx;
  float d2s = 0.0f;
  if (t < 32) {
    float q = cb[idx * 32 + t];
    qst[(size_t)row * 32 + t] = q;                 // q_st == q in forward value
    qb[(size_t)row * 32 + t]  = f32_to_bf16(q);
    float diff = q - se[t];
    d2s = diff * diff;
  }
  smin[t] = d2s;
  __syncthreads();
  for (int s = 16; s > 0; s >>= 1) {
    if (t < s) smin[t] += smin[t + s];
    __syncthreads();
  }
  if (t == 0) partial[row] = smin[0];
}

__global__ __launch_bounds__(256) void reduce_loss_kernel(
    const float* __restrict__ partial, int n, float scale, float* __restrict__ out)
{
  __shared__ float s[256];
  float acc = 0.0f;
  for (int i = threadIdx.x; i < n; i += 256) acc += partial[i];
  s[threadIdx.x] = acc;
  __syncthreads();
  for (int st = 128; st > 0; st >>= 1) {
    if (threadIdx.x < st) s[threadIdx.x] += s[threadIdx.x + st];
    __syncthreads();
  }
  if (threadIdx.x == 0) *out = s[0] * scale;
}

extern "C" void kernel_launch(void* const* d_in, const int* in_sizes, int n_in,
                              void* d_out, int out_size, void* d_ws, size_t ws_size,
                              hipStream_t stream) {
  (void)in_sizes; (void)n_in; (void)out_size; (void)ws_size;
  const float* x   = (const float*)d_in[0];
  const float* e1w = (const float*)d_in[1];
  const float* e1b = (const float*)d_in[2];
  const float* e2w = (const float*)d_in[3];
  const float* e2b = (const float*)d_in[4];
  const float* e3w = (const float*)d_in[5];
  const float* e3b = (const float*)d_in[6];
  const float* cb  = (const float*)d_in[7];
  const float* d1w = (const float*)d_in[8];
  const float* d1b = (const float*)d_in[9];
  const float* d2w = (const float*)d_in[10];
  const float* d2b = (const float*)d_in[11];
  const float* d3w = (const float*)d_in[12];
  const float* d3b = (const float*)d_in[13];

  float* recon = (float*)d_out;                       // [N, CIN]
  float* loss  = recon + (size_t)N_ROWS * CIN;        // [1]
  float* qst   = loss + 1;                            // [N, DLAT]

  char* ws = (char*)d_ws;
  size_t off = 0;
  auto alloc = [&](size_t bytes) -> char* {
    char* p = ws + off;
    off = (off + bytes + 255) & ~(size_t)255;
    return p;
  };
  unsigned short* xb   = (unsigned short*)alloc((size_t)N_ROWS * CIN * 2);
  unsigned short* w1t  = (unsigned short*)alloc((size_t)CIN * HID * 2);   // [HID,CIN]
  unsigned short* w2t  = (unsigned short*)alloc((size_t)HID * HID * 2);   // [HID,HID]
  unsigned short* w3t  = (unsigned short*)alloc((size_t)HID * DLAT * 2);  // [DLAT,HID]
  unsigned short* w4t  = (unsigned short*)alloc((size_t)DLAT * HID * 2);  // [HID,DLAT]
  unsigned short* w5t  = (unsigned short*)alloc((size_t)HID * HID * 2);   // [HID,HID]
  unsigned short* w6t  = (unsigned short*)alloc((size_t)HID * CIN * 2);   // [CIN,HID]
  unsigned short* buf1 = (unsigned short*)alloc((size_t)N_ROWS * HID * 2); // h1 / d1
  unsigned short* buf2 = (unsigned short*)alloc((size_t)N_ROWS * HID * 2); // h2 / d2
  float*          ef   = (float*)alloc((size_t)N_ROWS * DLAT * 4);
  unsigned short* qb   = (unsigned short*)alloc((size_t)N_ROWS * DLAT * 2);
  float*          part = (float*)alloc((size_t)N_ROWS * 4);

  // x -> bf16 (row-major is already what the A side wants)
  {
    size_t n = (size_t)N_ROWS * CIN;
    cvt_f32_bf16_kernel<<<dim3((unsigned)((n + 255) / 256)), dim3(256), 0, stream>>>(
        x, xb, (int)n);
  }
  // weights: f32 [K,N] -> bf16 [N,K]
  auto wtr = [&](const float* W, unsigned short* Wt, int K, int N) {
    transpose_f32_bf16_kernel<<<dim3(N / 32, K / 32), dim3(256), 0, stream>>>(W, Wt, K, N);
  };
  wtr(e1w, w1t, CIN, HID);
  wtr(e2w, w2t, HID, HID);
  wtr(e3w, w3t, HID, DLAT);
  wtr(d1w, w4t, DLAT, HID);
  wtr(d2w, w5t, HID, HID);
  wtr(d3w, w6t, HID, CIN);

  auto gemm = [&](const unsigned short* A, const unsigned short* Wt, const float* bias,
                  unsigned short* ob, float* of, int M, int N, int K, int act) {
    dim3 grid((N + 127) / 128, M / 256);
    gemm_bf16_wmma<<<grid, dim3(256), 0, stream>>>(A, Wt, bias, ob, of, M, N, K, act);
  };

  // encoder
  gemm(xb,   w1t, e1b, buf1, nullptr, N_ROWS, HID,  CIN, 0);
  gemm(buf1, w2t, e2b, buf2, nullptr, N_ROWS, HID,  HID, 0);
  gemm(buf2, w3t, e3b, nullptr, ef,   N_ROWS, DLAT, HID, 0);
  // vector quantization
  vq_kernel<<<dim3(N_ROWS), dim3(256), 0, stream>>>(ef, cb, qst, qb, part);
  reduce_loss_kernel<<<dim3(1), dim3(256), 0, stream>>>(
      part, N_ROWS, (1.0f + COMMITF) / ((float)N_ROWS * (float)DLAT), loss);
  // decoder
  gemm(qb,   w4t, d1b, buf1, nullptr, N_ROWS, HID, DLAT, 0);
  gemm(buf1, w5t, d2b, buf2, nullptr, N_ROWS, HID, HID,  0);
  gemm(buf2, w6t, d3b, nullptr, recon, N_ROWS, CIN, HID, 1);
}
